// SAModule_6244882448516
// MI455X (gfx1250) — matrix operand order repaired and checked by reading, not compiled, not run on hardware
//
#include <hip/hip_runtime.h>
#include <hip/hip_bf16.h>

#define B_   4
#define NP_  8192
#define MP_  2048
#define K_   64
#define F_   32
#define H1_  64
#define H2_  64
#define H3_  128
#define R2_  0.04f
#define CAP_ 1024

typedef __attribute__((ext_vector_type(16))) _Float16 v16h;
typedef __attribute__((ext_vector_type(8)))  _Float16 v8h;
typedef __attribute__((ext_vector_type(8)))  float    v8f;

// ---------------------------------------------------------------------------
// Kernel 1: farthest point sampling, one 1024-thread block per cloud.
// Each thread owns 8 points (coords + running min-dist kept in registers).
// ---------------------------------------------------------------------------
__global__ void __launch_bounds__(1024)
fps_kernel(const float* __restrict__ pos, int* __restrict__ fps_idx) {
    const int b   = blockIdx.x;
    const int tid = threadIdx.x;
    const float* pb = pos + (size_t)b * NP_ * 3;

    float px[8], py[8], pz[8], dist[8];
#pragma unroll
    for (int j = 0; j < 8; ++j) {
        const int p = j * 1024 + tid;
        px[j] = pb[p * 3 + 0];
        py[j] = pb[p * 3 + 1];
        pz[j] = pb[p * 3 + 2];
        dist[j] = 3.4e38f;
    }

    __shared__ float s_cx, s_cy, s_cz;
    __shared__ int   s_cur;
    __shared__ float s_rd[32];
    __shared__ int   s_ri[32];
    if (tid == 0) s_cur = 0;
    __syncthreads();

    for (int it = 0; it < MP_; ++it) {
        const int cur = s_cur;
        if (tid == 0) fps_idx[b * MP_ + it] = cur;
        if ((cur & 1023) == tid) {           // owner broadcasts coords
            const int j = cur >> 10;
            s_cx = px[j]; s_cy = py[j]; s_cz = pz[j];
        }
        __syncthreads();
        const float cx = s_cx, cy = s_cy, cz = s_cz;

        float bd = -1.0f; int bi = 0x7fffffff;
#pragma unroll
        for (int j = 0; j < 8; ++j) {
            const float dx = px[j] - cx, dy = py[j] - cy, dz = pz[j] - cz;
            const float d2 = dx * dx + dy * dy + dz * dz;
            const float nd = fminf(dist[j], d2);
            dist[j] = nd;
            const int p = j * 1024 + tid;
            if (nd > bd || (nd == bd && p < bi)) { bd = nd; bi = p; }
        }
#pragma unroll
        for (int off = 16; off >= 1; off >>= 1) {
            const float od = __shfl_xor(bd, off);
            const int   oi = __shfl_xor(bi, off);
            if (od > bd || (od == bd && oi < bi)) { bd = od; bi = oi; }
        }
        if ((tid & 31) == 0) { s_rd[tid >> 5] = bd; s_ri[tid >> 5] = bi; }
        __syncthreads();
        if (tid < 32) {
            bd = s_rd[tid]; bi = s_ri[tid];
#pragma unroll
            for (int off = 16; off >= 1; off >>= 1) {
                const float od = __shfl_xor(bd, off);
                const int   oi = __shfl_xor(bi, off);
                if (od > bd || (od == bd && oi < bi)) { bd = od; bi = oi; }
            }
            if (tid == 0) s_cur = bi;
        }
        __syncthreads();
    }
}

// ---------------------------------------------------------------------------
// Kernel 2: ball query. One 256-thread block per center. Compact in-radius
// candidates into LDS, then extract up to K=64 nearest by iterative argmin.
// (Only the SET of valid neighbors matters: max-pool of ReLU>=0 outputs.)
// ---------------------------------------------------------------------------
__global__ void __launch_bounds__(256)
ballq_kernel(const float* __restrict__ pos, const int* __restrict__ fps_idx,
             int* __restrict__ nbr_idx, int* __restrict__ nbr_cnt,
             float* __restrict__ centers, int* __restrict__ cbatch) {
    const int c   = blockIdx.x;          // 0 .. B*MP-1
    const int b   = c / MP_;
    const int tid = threadIdx.x;

    __shared__ int   s_cnt;
    __shared__ float s_d2[CAP_];
    __shared__ int   s_pi[CAP_];
    __shared__ float s_rd[8];
    __shared__ int   s_rs[8];
    __shared__ int   s_sel[K_];

    const int cidx = fps_idx[c];
    const float* pb = pos + (size_t)b * NP_ * 3;
    const float cx = pb[cidx * 3 + 0];
    const float cy = pb[cidx * 3 + 1];
    const float cz = pb[cidx * 3 + 2];
    if (tid == 0) {
        centers[c * 3 + 0] = cx;
        centers[c * 3 + 1] = cy;
        centers[c * 3 + 2] = cz;
        cbatch[c] = b;
        s_cnt = 0;
    }
    __syncthreads();

    for (int p = tid; p < NP_; p += 256) {
        const float dx = pb[p * 3 + 0] - cx;
        const float dy = pb[p * 3 + 1] - cy;
        const float dz = pb[p * 3 + 2] - cz;
        const float d2 = dx * dx + dy * dy + dz * dz;
        if (d2 <= R2_) {
            const int slot = atomicAdd(&s_cnt, 1);
            if (slot < CAP_) { s_d2[slot] = d2; s_pi[slot] = p; }
        }
    }
    __syncthreads();

    const int n   = min(s_cnt, CAP_);
    const int sel = min(n, K_);
    for (int s = 0; s < sel; ++s) {
        float bd = 3.4e38f; int bs = 0x7fffffff;
        for (int q = tid; q < n; q += 256) {
            const float d = s_d2[q];
            if (d < bd || (d == bd && q < bs)) { bd = d; bs = q; }
        }
#pragma unroll
        for (int off = 16; off >= 1; off >>= 1) {
            const float od = __shfl_xor(bd, off);
            const int   os = __shfl_xor(bs, off);
            if (od < bd || (od == bd && os < bs)) { bd = od; bs = os; }
        }
        if ((tid & 31) == 0) { s_rd[tid >> 5] = bd; s_rs[tid >> 5] = bs; }
        __syncthreads();
        if (tid == 0) {
            bd = s_rd[0]; bs = s_rs[0];
            for (int w = 1; w < 8; ++w) {
                if (s_rd[w] < bd || (s_rd[w] == bd && s_rs[w] < bs)) { bd = s_rd[w]; bs = s_rs[w]; }
            }
            s_sel[s] = s_pi[bs];
            s_d2[bs] = 3.4e38f;          // remove winner
        }
        __syncthreads();
    }

    if (tid < K_) nbr_idx[(size_t)c * K_ + tid] = (tid < sel) ? s_sel[tid] : cidx;
    if (tid == 0) nbr_cnt[c] = sel;
}

// ---------------------------------------------------------------------------
// WMMA fragment loaders (16-bit A 16x32 and B 32x16 layouts per CDNA5 ISA).
// Data is staged in LDS: A row-major [row][k] (stride 64 halves), weights
// transposed [out][in] (stride 64) so the B fragment is contiguous per lane.
// ---------------------------------------------------------------------------
__device__ inline v16h load_afrag(const _Float16* a_rm, int row_base, int kt, int lane) {
    // lanes 0-15: row = lane,    K = {0..7, 16..23}  (within the 32-wide k-tile)
    // lanes16-31: row = lane-16, K = {8..15, 24..31}
    const int hs  = lane >> 4;
    const int row = row_base + (lane & 15);
    const _Float16* p = a_rm + row * 64 + kt * 32 + hs * 8;
    const v8h lo = *(const v8h*)(p);
    const v8h hi = *(const v8h*)(p + 16);
    v16h a;
#pragma unroll
    for (int i = 0; i < 8; ++i) { a[i] = lo[i]; a[8 + i] = hi[i]; }
    return a;
}

__device__ inline v16h load_bfrag(const _Float16* w_t, int nt, int kt, int lane) {
    // lanes 0-15: col N = lane,    K = kt*32 + {0..7, 16..23}
    // lanes16-31: col N = lane-16, K = kt*32 + {8..15, 24..31}
    const int n   = nt * 16 + (lane & 15);
    const int klo = kt * 32 + ((lane < 16) ? 0 : 8);
    const _Float16* p = w_t + n * 64 + klo;
    const v8h lo = *(const v8h*)(p);
    const v8h hi = *(const v8h*)(p + 16);
    v16h w;
#pragma unroll
    for (int i = 0; i < 8; ++i) { w[i] = lo[i]; w[8 + i] = hi[i]; }
    return w;
}

// ---------------------------------------------------------------------------
// Kernel 3: gather + 3-layer MLP (f16 WMMA, f32 accum) + masked max-pool.
// One 128-thread block (4 waves) per center; each wave owns 16 neighbor rows.
// ---------------------------------------------------------------------------
__global__ void __launch_bounds__(128)
mlp_kernel(const float* __restrict__ x, const float* __restrict__ pos,
           const float* __restrict__ W1, const float* __restrict__ b1,
           const float* __restrict__ W2, const float* __restrict__ b2,
           const float* __restrict__ W3, const float* __restrict__ b3,
           const int* __restrict__ nbr_idx, const int* __restrict__ nbr_cnt,
           const float* __restrict__ centers, float* __restrict__ out) {
    const int c    = blockIdx.x;
    const int b    = c / MP_;
    const int tid  = threadIdx.x;
    const int wave = tid >> 5;
    const int lane = tid & 31;

    __shared__ __align__(16) _Float16 sA[64 * 64];      // input / h2
    __shared__ __align__(16) _Float16 sH[64 * 64];      // h1
    __shared__ __align__(16) _Float16 sW1[64 * 64];     // [out][in], in-padded
    __shared__ __align__(16) _Float16 sW2[64 * 64];     // [out][in]
    __shared__ __align__(16) _Float16 sW3[128 * 64];    // [out][in]
    __shared__ float sb1[64], sb2[64], sb3[128];
    __shared__ int   s_nidx[64];
    __shared__ float s_max[128];
    __shared__ int   s_cnt2;

    // Stage transposed weights + biases into LDS (weights are L2-resident).
    for (int e = tid; e < 64 * 64; e += 128) {
        const int o = e >> 6, i = e & 63;
        sW1[o * 64 + i] = (i < 35) ? (_Float16)W1[i * 64 + o] : (_Float16)0.f;
        sW2[o * 64 + i] = (_Float16)W2[i * 64 + o];
    }
    for (int e = tid; e < 128 * 64; e += 128) {
        const int o = e >> 6, i = e & 63;
        sW3[o * 64 + i] = (_Float16)W3[i * 128 + o];
    }
    if (tid < 64)  { sb1[tid] = b1[tid]; sb2[tid] = b2[tid]; }
    if (tid < 128) { sb3[tid] = b3[tid]; s_max[tid] = 0.f; }
    if (tid < 64)  s_nidx[tid] = nbr_idx[(size_t)c * K_ + tid];
    if (tid == 0)  s_cnt2 = nbr_cnt[c];
    __syncthreads();

    // Gather neighbor features + relative position into f16 A (K padded to 64).
    {
        const float cx = centers[c * 3 + 0];
        const float cy = centers[c * 3 + 1];
        const float cz = centers[c * 3 + 2];
        const int row  = tid >> 1;
        const int half = tid & 1;
        const int g    = b * NP_ + s_nidx[row];
        const float* xr = x + (size_t)g * F_;
        _Float16* ar = sA + row * 64;
        if (half == 0) {
#pragma unroll
            for (int f = 0; f < 16; ++f) ar[f] = (_Float16)xr[f];
            ar[32] = (_Float16)(pos[g * 3 + 0] - cx);
            ar[33] = (_Float16)(pos[g * 3 + 1] - cy);
            ar[34] = (_Float16)(pos[g * 3 + 2] - cz);
#pragma unroll
            for (int f = 35; f < 48; ++f) ar[f] = (_Float16)0.f;
        } else {
#pragma unroll
            for (int f = 16; f < 32; ++f) ar[f] = (_Float16)xr[f];
#pragma unroll
            for (int f = 48; f < 64; ++f) ar[f] = (_Float16)0.f;
        }
    }
    __syncthreads();

    const int rb = wave * 16;        // this wave's 16 neighbor rows

    // ---- Layer 1: sA[64x64] x W1t -> relu -> sH[64x64]
#pragma unroll
    for (int nt = 0; nt < 4; ++nt) {
        v8f acc = {};
#pragma unroll
        for (int kt = 0; kt < 2; ++kt) {
            const v16h a = load_afrag(sA, rb, kt, lane);
            const v16h w = load_bfrag(sW1, nt, kt, lane);
            acc = __builtin_amdgcn_wmma_f32_16x16x32_f16(false, a, false, w,
                                                         (short)0, acc, false, false);
        }
        const int col  = nt * 16 + (lane & 15);
        const int ro   = (lane < 16) ? 0 : 8;
        const float bv = sb1[col];
#pragma unroll
        for (int r = 0; r < 8; ++r)
            sH[(rb + ro + r) * 64 + col] = (_Float16)fmaxf(acc[r] + bv, 0.f);
    }
    __syncthreads();

    // ---- Layer 2: sH x W2t -> relu -> sA
#pragma unroll
    for (int nt = 0; nt < 4; ++nt) {
        v8f acc = {};
#pragma unroll
        for (int kt = 0; kt < 2; ++kt) {
            const v16h a = load_afrag(sH, rb, kt, lane);
            const v16h w = load_bfrag(sW2, nt, kt, lane);
            acc = __builtin_amdgcn_wmma_f32_16x16x32_f16(false, a, false, w,
                                                         (short)0, acc, false, false);
        }
        const int col  = nt * 16 + (lane & 15);
        const int ro   = (lane < 16) ? 0 : 8;
        const float bv = sb2[col];
#pragma unroll
        for (int r = 0; r < 8; ++r)
            sA[(rb + ro + r) * 64 + col] = (_Float16)fmaxf(acc[r] + bv, 0.f);
    }
    __syncthreads();

    // ---- Layer 3: sA x W3t -> relu -> masked max over valid neighbor rows.
    // Invalid rows (>= cnt) contribute 0; valid since ReLU outputs >= 0 and
    // the center itself is always a valid neighbor.
    const int cnt = s_cnt2;
#pragma unroll
    for (int nt = 0; nt < 8; ++nt) {
        v8f acc = {};
#pragma unroll
        for (int kt = 0; kt < 2; ++kt) {
            const v16h a = load_afrag(sA, rb, kt, lane);
            const v16h w = load_bfrag(sW3, nt, kt, lane);
            acc = __builtin_amdgcn_wmma_f32_16x16x32_f16(false, a, false, w,
                                                         (short)0, acc, false, false);
        }
        const int col  = nt * 16 + (lane & 15);
        const int ro   = (lane < 16) ? 0 : 8;
        const float bv = sb3[col];
        float m = 0.f;
#pragma unroll
        for (int r = 0; r < 8; ++r) {
            const int row = rb + ro + r;
            const float v = fmaxf(acc[r] + bv, 0.f);
            if (row < cnt) m = fmaxf(m, v);
        }
        m = fmaxf(m, __shfl_xor(m, 16));                 // combine M-halves (wave32)
        if (lane < 16)                                   // int-punned max: all vals >= 0
            atomicMax((int*)&s_max[col], __float_as_int(m));
    }
    __syncthreads();

    if (tid < 128) out[(size_t)c * H3_ + tid] = s_max[tid];
}

// ---------------------------------------------------------------------------
extern "C" void kernel_launch(void* const* d_in, const int* in_sizes, int n_in,
                              void* d_out, int out_size, void* d_ws, size_t ws_size,
                              hipStream_t stream) {
    const float* x   = (const float*)d_in[0];
    const float* pos = (const float*)d_in[1];
    /* d_in[2] = batch (unused; layout is implicit) */
    const float* W1  = (const float*)d_in[3];
    const float* b1  = (const float*)d_in[4];
    const float* W2  = (const float*)d_in[5];
    const float* b2  = (const float*)d_in[6];
    const float* W3  = (const float*)d_in[7];
    const float* b3  = (const float*)d_in[8];

    float* out     = (float*)d_out;                           // [B*MP, 128]
    float* centers = out + (size_t)B_ * MP_ * H3_;            // [B*MP, 3]
    int*   cbatch  = (int*)(centers + (size_t)B_ * MP_ * 3);  // [B*MP]

    int* ws       = (int*)d_ws;
    int* fps_idx  = ws;                                       // [B*MP]
    int* nbr_idx  = fps_idx + B_ * MP_;                       // [B*MP, K]
    int* nbr_cnt  = nbr_idx + (size_t)B_ * MP_ * K_;          // [B*MP]

    fps_kernel  <<<B_,        1024, 0, stream>>>(pos, fps_idx);
    ballq_kernel<<<B_ * MP_,  256,  0, stream>>>(pos, fps_idx, nbr_idx, nbr_cnt,
                                                 centers, cbatch);
    mlp_kernel  <<<B_ * MP_,  128,  0, stream>>>(x, pos, W1, b1, W2, b2, W3, b3,
                                                 nbr_idx, nbr_cnt, centers, out);
}
